// CrossAttention_30176440221764
// MI455X (gfx1250) — matrix operand order repaired and checked by reading, not compiled, not run on hardware
//
#include <hip/hip_runtime.h>

// ---------------- problem constants ----------------
#define BB   4
#define SQ_  2048
#define SKV_ 2048
#define DE_  1024
#define DC_  768
#define DA_  1024
#define NH_  16
#define DH_  64

typedef unsigned short u16t;
typedef __attribute__((ext_vector_type(16))) __bf16          bf16x16;
typedef __attribute__((ext_vector_type(8)))  float           f32x8;
typedef __attribute__((ext_vector_type(4)))  unsigned int    u32x4;
typedef __attribute__((ext_vector_type(16))) unsigned short  u16x16;
typedef int i32x4 __attribute__((vector_size(16)));   // matches async-LDS builtin param

#define AS1 __attribute__((address_space(1)))
#define AS3 __attribute__((address_space(3)))

#if __has_builtin(__builtin_amdgcn_global_load_async_to_lds_b128)
#define HAVE_ASYNC_LDS 1
#else
#define HAVE_ASYNC_LDS 0
#endif

#if __has_builtin(__builtin_amdgcn_ds_swizzle)
#define HAVE_SWZ 1
#else
#define HAVE_SWZ 0
#endif

struct __attribute__((aligned(16))) Frag32B { u32x4 lo, hi; };

__device__ __forceinline__ u16t f2bf(float f) {
  unsigned int u = __builtin_bit_cast(unsigned int, f);
  u += 0x7FFFu + ((u >> 16) & 1u);        // round-to-nearest-even
  return (u16t)(u >> 16);
}

// load a 16-element bf16 fragment as two 16B chunks (global or LDS)
__device__ __forceinline__ bf16x16 ldfrag(const u16t* p0, const u16t* p1) {
  Frag32B t;
  t.lo = *(const u32x4*)p0;
  t.hi = *(const u32x4*)p1;
  return __builtin_bit_cast(bf16x16, t);
}

__device__ __forceinline__ f32x8 wmma_bf16(bf16x16 a, bf16x16 b, f32x8 c) {
  return __builtin_amdgcn_wmma_f32_16x16x32_bf16(false, a, false, b, (short)0, c,
                                                 false, false);
}

__device__ __forceinline__ void wait_async_lds() {
#if HAVE_ASYNC_LDS
#if __has_builtin(__builtin_amdgcn_s_wait_asynccnt)
  __builtin_amdgcn_s_wait_asynccnt(0);
#else
  asm volatile("s_wait_asynccnt 0" ::: "memory");
#endif
#endif
}

// max-reduce across each 16-lane half (butterfly over xor masks 1,2,4,8)
__device__ __forceinline__ float red_max16(float m) {
#if HAVE_SWZ
  m = fmaxf(m, __builtin_bit_cast(float, __builtin_amdgcn_ds_swizzle(
                   __builtin_bit_cast(int, m), 0x041F)));   // SWAPX1
  m = fmaxf(m, __builtin_bit_cast(float, __builtin_amdgcn_ds_swizzle(
                   __builtin_bit_cast(int, m), 0x081F)));   // SWAPX2
  m = fmaxf(m, __builtin_bit_cast(float, __builtin_amdgcn_ds_swizzle(
                   __builtin_bit_cast(int, m), 0x101F)));   // SWAPX4
  m = fmaxf(m, __builtin_bit_cast(float, __builtin_amdgcn_ds_swizzle(
                   __builtin_bit_cast(int, m), 0x201F)));   // SWAPX8
#else
  #pragma unroll
  for (int off = 1; off < 16; off <<= 1) m = fmaxf(m, __shfl_xor(m, off, 32));
#endif
  return m;
}

// ---------------- elementwise conversions ----------------
__global__ void k_cvt_bf16(const float* __restrict__ in, u16t* __restrict__ out, int n) {
  int i = blockIdx.x * blockDim.x + threadIdx.x;
  if (i < n) out[i] = f2bf(in[i]);
}

// W (K x N) fp32 row-major  ->  Wt (N x K) bf16 row-major
__global__ void k_cvt_transpose(const float* __restrict__ W, u16t* __restrict__ Wt,
                                int K, int N) {
  int i = blockIdx.x * blockDim.x + threadIdx.x;
  if (i < K * N) {
    int k = i / N, n = i % N;
    Wt[(size_t)n * K + k] = f2bf(W[i]);
  }
}

// ---------------- bf16 WMMA GEMM:  C = oscale * (A(MxK) * Wt(NxK)^T + bias) ----------------
// wave computes a 32x64 tile (2 row-tiles share 4 B fragments); block = 8 waves
template <bool OUT_F32>
__global__ __launch_bounds__(256) void k_gemm(const u16t* __restrict__ A,
                                              const u16t* __restrict__ Wt,
                                              const float* __restrict__ bias,
                                              void* __restrict__ out,
                                              int M, int N, int K, float oscale) {
  const int lane = threadIdx.x & 31;
  const int wave = threadIdx.x >> 5;
  const int tile = blockIdx.x * 8 + wave;
  const int colTiles = N >> 6;
  const int row0 = (tile / colTiles) << 5;   // 32 rows per wave
  const int col0 = (tile % colTiles) << 6;   // 64 cols per wave
  const int ln   = lane & 15;
  const int hi16 = lane >> 4;
  const int kbA  = hi16 ? 8 : 0;    // A-frag K interleave {0-7,16-23}/{8-15,24-31}
  const int kbB  = hi16 ? 8 : 0;    // B-frag K interleave {0-7,16-23}/{8-15,24-31}

  const u16t* arow0 = A + (size_t)(row0 + ln) * K;
  const u16t* arow1 = arow0 + (size_t)16 * K;

  f32x8 acc[2][4];
  #pragma unroll
  for (int j = 0; j < 4; ++j) {
    float bv = bias[col0 + j * 16 + ln];
    #pragma unroll
    for (int r = 0; r < 8; ++r) { acc[0][j][r] = bv; acc[1][j][r] = bv; }
  }

  for (int k0 = 0; k0 < K; k0 += 32) {
    bf16x16 a0 = ldfrag(arow0 + k0 + kbA, arow0 + k0 + kbA + 16);
    bf16x16 a1 = ldfrag(arow1 + k0 + kbA, arow1 + k0 + kbA + 16);
    if (k0 + 32 < K) {
      __builtin_prefetch(arow0 + k0 + 32, 0, 1);
      __builtin_prefetch(arow1 + k0 + 32, 0, 1);
    }
    #pragma unroll
    for (int j = 0; j < 4; ++j) {
      const u16t* wrow = Wt + (size_t)(col0 + j * 16 + ln) * K + k0 + kbB;
      bf16x16 b = ldfrag(wrow, wrow + 16);
      acc[0][j] = wmma_bf16(a0, b, acc[0][j]);
      acc[1][j] = wmma_bf16(a1, b, acc[1][j]);
    }
  }

  #pragma unroll
  for (int i = 0; i < 2; ++i) {
    #pragma unroll
    for (int j = 0; j < 4; ++j) {
      int col = col0 + j * 16 + ln;
      #pragma unroll
      for (int r = 0; r < 8; ++r) {
        int row = row0 + i * 16 + r + hi16 * 8;
        float v = acc[i][j][r] * oscale;
        if (OUT_F32) ((float*)out)[(size_t)row * N + col] = v;
        else         ((u16t*)out)[(size_t)row * N + col]  = f2bf(v);
      }
    }
  }
}

// ---------------- flash attention: block = (b, h, 128 q rows), 8 waves x 16 rows ----------------
// Q is pre-scaled by 1/sqrt(d_head) in its projection, so scores need no scaling here.
__global__ __launch_bounds__(256) void k_attn(const u16t* __restrict__ Q,
                                              const u16t* __restrict__ Kb,
                                              const u16t* __restrict__ Vb,
                                              u16t* __restrict__ Ob) {
  __shared__ u16t ldsK[32 * 64];       // K tile row-major: 32 keys x 64 d
  __shared__ u16t ldsVt[64 * 32];      // V tile TRANSPOSED: 64 d x 32 keys
  __shared__ u16t ldsP[8][16 * 32];    // per-wave P transpose buffer

  const int tid  = threadIdx.x;
  const int lane = tid & 31;
  const int wave = tid >> 5;
  const int b    = blockIdx.z;
  const int h    = blockIdx.y;
  const int q0   = blockIdx.x * 128 + wave * 16;
  const int ln   = lane & 15;
  const int hi16 = lane >> 4;
  const int kbA  = hi16 ? 8 : 0;
  const int kbB  = hi16 ? 8 : 0;

  const size_t rs = DA_;  // row stride (16 heads * 64)

  // Q fragments for this wave's 16 rows (d = 0..31 and 32..63)
  const u16t* qrow = Q + ((size_t)b * SQ_ + q0 + ln) * rs + h * DH_;
  bf16x16 aq0 = ldfrag(qrow + kbA,      qrow + kbA + 16);
  bf16x16 aq1 = ldfrag(qrow + 32 + kbA, qrow + 32 + kbA + 16);

  const u16t* kbase = Kb + ((size_t)b * SKV_) * rs + h * DH_;
  const u16t* vbase = Vb + ((size_t)b * SKV_) * rs + h * DH_;

  // all-ones B fragment: row sums via an extra WMMA (P @ 1)
  bf16x16 bones;
  {
    u16x16 t;
    #pragma unroll
    for (int e = 0; e < 16; ++e) t[e] = 0x3F80;  // bf16(1.0)
    bones = __builtin_bit_cast(bf16x16, t);
  }

  float rowmax[8];
  f32x8 o[4], osum;
  #pragma unroll
  for (int r = 0; r < 8; ++r) { rowmax[r] = -1e30f; osum[r] = 0.0f; }
  #pragma unroll
  for (int j = 0; j < 4; ++j)
    #pragma unroll
    for (int r = 0; r < 8; ++r) o[j][r] = 0.0f;

  const int rowi = tid >> 3, c8 = (tid & 7) << 3;

  for (int kv0 = 0; kv0 < SKV_; kv0 += 32) {
    __syncthreads();  // protect prior-iteration LDS reads
    {   // stage K (async direct-to-LDS) and V (transposed through registers)
      const u16t* gk = kbase + (size_t)(kv0 + rowi) * rs + c8;
      const u16t* gv = vbase + (size_t)(kv0 + rowi) * rs + c8;
#if HAVE_ASYNC_LDS
      __builtin_amdgcn_global_load_async_to_lds_b128(
          (AS1 i32x4*)gk, (AS3 i32x4*)&ldsK[rowi * 64 + c8], 0, 0);
#else
      *(u32x4*)&ldsK[rowi * 64 + c8] = *(const u32x4*)gk;
#endif
      u32x4 vv = *(const u32x4*)gv;
      u16t vs[8];
      *(u32x4*)vs = vv;
      #pragma unroll
      for (int e = 0; e < 8; ++e)
        ldsVt[(c8 + e) * 32 + rowi] = vs[e];      // Vt[d][kv]
      if (kv0 + 32 < SKV_) {
        __builtin_prefetch(gk + 32 * rs, 0, 1);
        __builtin_prefetch(gv + 32 * rs, 0, 1);
      }
    }
    wait_async_lds();
    __syncthreads();

    // scores: S = Qs * K^T for 16 q x 32 kv (two 16x16 tiles)
    f32x8 s0, s1;
    #pragma unroll
    for (int r = 0; r < 8; ++r) { s0[r] = 0.0f; s1[r] = 0.0f; }
    {
      const u16t* kr0 = &ldsK[ln * 64 + kbB];          // kv col = ln
      const u16t* kr1 = &ldsK[(16 + ln) * 64 + kbB];   // kv col = 16 + ln
      s0 = wmma_bf16(aq0, ldfrag(kr0,      kr0 + 16), s0);
      s0 = wmma_bf16(aq1, ldfrag(kr0 + 32, kr0 + 48), s0);
      s1 = wmma_bf16(aq0, ldfrag(kr1,      kr1 + 16), s1);
      s1 = wmma_bf16(aq1, ldfrag(kr1 + 32, kr1 + 48), s1);
    }

    // online softmax: running max (ds_swizzle butterfly) + rescale accumulators
    float p0[8], p1[8];
    #pragma unroll
    for (int r = 0; r < 8; ++r) {
      float m = red_max16(fmaxf(s0[r], s1[r]));
      float nm = fmaxf(rowmax[r], m);
      float corr = __expf(rowmax[r] - nm);
      rowmax[r] = nm;
      p0[r] = __expf(s0[r] - nm);
      p1[r] = __expf(s1[r] - nm);
      osum[r] *= corr;
      #pragma unroll
      for (int j = 0; j < 4; ++j) o[j][r] *= corr;
    }

    // transpose P through per-wave LDS into an A-fragment
    u16t* pw = ldsP[wave];
    #pragma unroll
    for (int r = 0; r < 8; ++r) {
      int row = r + hi16 * 8;
      pw[row * 32 + ln]      = f2bf(p0[r]);
      pw[row * 32 + 16 + ln] = f2bf(p1[r]);
    }
    asm volatile("s_wait_dscnt 0" ::: "memory");   // wave-internal LDS RAW ordering
    bf16x16 ap = ldfrag(pw + ln * 32 + kbA, pw + ln * 32 + kbA + 16);

    // O += P * V  (4 d-tiles, contiguous b128 reads from transposed V)
    #pragma unroll
    for (int j = 0; j < 4; ++j) {
      const u16t* vr = &ldsVt[(j * 16 + ln) * 32 + kbB];
      o[j] = wmma_bf16(ap, ldfrag(vr, vr + 16), o[j]);
    }
    // row sums: osum += P @ ones (replaces a second shuffle reduction)
    osum = wmma_bf16(ap, bones, osum);
  }

  // normalize and store context (bf16, row-major (b, q, h*64+d))
  u16t* obase = Ob + ((size_t)b * SQ_ + q0) * rs + h * DH_;
  #pragma unroll
  for (int r = 0; r < 8; ++r) {
    float inv = 1.0f / osum[r];
    int row = r + hi16 * 8;
    #pragma unroll
    for (int j = 0; j < 4; ++j)
      obase[(size_t)row * rs + j * 16 + ln] = f2bf(o[j][r] * inv);
  }
}

// ---------------- host-side orchestration ----------------
extern "C" void kernel_launch(void* const* d_in, const int* in_sizes, int n_in,
                              void* d_out, int out_size, void* d_ws, size_t ws_size,
                              hipStream_t stream) {
  (void)in_sizes; (void)n_in; (void)out_size; (void)ws_size;
  const float* x  = (const float*)d_in[0];
  const float* y  = (const float*)d_in[1];
  const float* Wq = (const float*)d_in[2];
  const float* bq = (const float*)d_in[3];
  const float* Wk = (const float*)d_in[4];
  const float* bk = (const float*)d_in[5];
  const float* Wv = (const float*)d_in[6];
  const float* bv = (const float*)d_in[7];
  const float* Wo = (const float*)d_in[8];
  const float* bo = (const float*)d_in[9];

  const size_t nX = (size_t)BB * SQ_ * DE_;    // 8.39M
  const size_t nY = (size_t)BB * SKV_ * DC_;   // 6.29M
  const size_t nQ = (size_t)BB * SQ_ * DA_;    // 8.39M

  char* ws = (char*)d_ws;
  size_t off = 0;
  auto alloc = [&](size_t bytes) -> char* {
    char* p = ws + off;
    off += (bytes + 255) & ~(size_t)255;
    return p;
  };
  u16t* xh  = (u16t*)alloc(nX * 2);
  u16t* yh  = (u16t*)alloc(nY * 2);
  u16t* WqT = (u16t*)alloc((size_t)DE_ * DA_ * 2);
  u16t* WkT = (u16t*)alloc((size_t)DC_ * DA_ * 2);
  u16t* WvT = (u16t*)alloc((size_t)DC_ * DA_ * 2);
  u16t* WoT = (u16t*)alloc((size_t)DA_ * DE_ * 2);
  u16t* Qb  = (u16t*)alloc(nQ * 2);
  u16t* Kb  = (u16t*)alloc(nQ * 2);
  u16t* Vb  = (u16t*)alloc(nQ * 2);
  u16t* Ab  = xh;  // attention output aliases xh (dead after Q projection)

  const int M = BB * SQ_;  // 8192 rows for all GEMMs

  k_cvt_bf16<<<(int)(nX / 256), 256, 0, stream>>>(x, xh, (int)nX);
  k_cvt_bf16<<<(int)(nY / 256), 256, 0, stream>>>(y, yh, (int)nY);
  k_cvt_transpose<<<(DE_ * DA_) / 256, 256, 0, stream>>>(Wq, WqT, DE_, DA_);
  k_cvt_transpose<<<(DC_ * DA_) / 256, 256, 0, stream>>>(Wk, WkT, DC_, DA_);
  k_cvt_transpose<<<(DC_ * DA_) / 256, 256, 0, stream>>>(Wv, WvT, DC_, DA_);
  k_cvt_transpose<<<(DA_ * DE_) / 256, 256, 0, stream>>>(Wo, WoT, DA_, DE_);

  const int gemmBlocks = (M / 32) * (DA_ / 64) / 8;  // 512
  // Q projection pre-scaled by 1/sqrt(d_head): scores = (q/8) . k
  k_gemm<false><<<gemmBlocks, 256, 0, stream>>>(xh, WqT, bq, Qb, M, DA_, DE_, 0.125f);
  k_gemm<false><<<gemmBlocks, 256, 0, stream>>>(yh, WkT, bk, Kb, M, DA_, DC_, 1.0f);
  k_gemm<false><<<gemmBlocks, 256, 0, stream>>>(yh, WvT, bv, Vb, M, DA_, DC_, 1.0f);

  dim3 agrid(SQ_ / 128, NH_, BB);  // (16, 16, 4)
  k_attn<<<agrid, 256, 0, stream>>>(Qb, Kb, Vb, Ab);

  k_gemm<true><<<gemmBlocks, 256, 0, stream>>>(Ab, WoT, bo, d_out, M, DE_, DA_, 1.0f);
}
